// EdgeDyn_64158221468180
// MI455X (gfx1250) — matrix operand order repaired and checked by reading, not compiled, not run on hardware
//
#include <hip/hip_runtime.h>
#include <hip/hip_bf16.h>

#define N_NODES  50000
#define N_EDGES  800000
#define DDIM     64
#define HDIM     128
#define NTILES   (N_NODES / 16)        // 3125 exactly
#define DT_F     0.1f
#define INV_TAU  0.5f                  // 1/TAU, TAU=2.0
#define LAMBDA_W 1e-4f

typedef __attribute__((ext_vector_type(16))) __bf16 bf16x16;
typedef __attribute__((ext_vector_type(8)))  float  f32x8;

union Frag16 { bf16x16 v; unsigned int u[8]; };

// float -> bf16 (round to nearest even on the mantissa boundary)
__device__ __forceinline__ unsigned short f2bf(float f) {
    unsigned int u = __float_as_uint(f);
    u += 0x7FFFu + ((u >> 16) & 1u);
    return (unsigned short)(u >> 16);
}

// -------------------------------------------------------------------------
// Zero a float buffer (n multiple of 4)
__global__ void zero_f32(float* __restrict__ p, int n) {
    int i = (blockIdx.x * blockDim.x + threadIdx.x) * 4;
    if (i < n) *(float4*)(p + i) = make_float4(0.f, 0.f, 0.f, 0.f);
}

// -------------------------------------------------------------------------
// Pre-swizzle W1 (128x128) and W2 (128x64) into WMMA B-fragment order, bf16.
// Tile layout: [(kc*NT + nt)*32 + lane]*16 + e ; K = kc*32 + 8*(lane>=16) + e (e<8), kc*32 + 16 + 8*(lane>=16) + (e-8) (e>=8)
__global__ void prep_weights(const float* __restrict__ W1, const float* __restrict__ W2,
                             unsigned short* __restrict__ W1bf, unsigned short* __restrict__ W2bf) {
    int t = blockIdx.x * blockDim.x + threadIdx.x;
    if (t < 4 * 8 * 512) {                       // W1: 4 k-chunks x 8 n-tiles
        int tile = t >> 9, r = t & 511;
        int lane = r >> 4, e = r & 15;
        int kc = tile >> 3, nt = tile & 7;
        int gk = kc * 32 + ((lane >> 4) * 8) + (e < 8 ? e : 16 + (e - 8));
        int gn = nt * 16 + (lane & 15);
        W1bf[t] = f2bf(W1[gk * HDIM + gn]);
    } else if (t < 4 * 8 * 512 + 4 * 4 * 512) {  // W2: 4 k-chunks x 4 n-tiles
        int t2 = t - 4 * 8 * 512;
        int tile = t2 >> 9, r = t2 & 511;
        int lane = r >> 4, e = r & 15;
        int kc = tile >> 2, nt = tile & 3;
        int gk = kc * 32 + ((lane >> 4) * 8) + (e < 8 ? e : 16 + (e - 8));
        int gn = nt * 16 + (lane & 15);
        W2bf[t2] = f2bf(W2[gk * DDIM + gn]);
    }
}

// -------------------------------------------------------------------------
// messages[tgt] += w * x[src] : 16 lanes per edge, float4 per lane, L2 atomics
__global__ void scatter_msgs(const int* __restrict__ ei, const float* __restrict__ w,
                             const float* __restrict__ x, float* __restrict__ msg) {
    int t = blockIdx.x * blockDim.x + threadIdx.x;
    if (t >= N_EDGES * 16) return;
    int eid = t >> 4, part = t & 15;
    int src = ei[eid];
    int tgt = ei[N_EDGES + eid];
    float wv = w[eid];
    float4 xs = *(const float4*)(x + src * DDIM + part * 4);
    float* mp = msg + tgt * DDIM + part * 4;
    atomicAdd(mp + 0, wv * xs.x);
    atomicAdd(mp + 1, wv * xs.y);
    atomicAdd(mp + 2, wv * xs.z);
    atomicAdd(mp + 3, wv * xs.w);
}

// -------------------------------------------------------------------------
// MLP + x/e integration. One wave = one 16-node tile. bf16 WMMA, f32 accum.
__global__ __launch_bounds__(256)
void node_mlp(float* __restrict__ x, float* __restrict__ e, const float* __restrict__ msg,
              const unsigned short* __restrict__ W1bf, const unsigned short* __restrict__ W2bf,
              const float* __restrict__ b1, const float* __restrict__ b2) {
    __shared__ unsigned short lds[8][16 * HDIM];   // 4KB bf16 tile per wave
    const int wave = threadIdx.x >> 5;
    const int lane = threadIdx.x & 31;
    int tile = blockIdx.x * 8 + wave;
    const bool active = (tile < NTILES);
    if (!active) tile = NTILES - 1;                // keep all waves/lanes live (EXEC all-1s)
    const int base = tile * 16;
    unsigned short* h = lds[wave];

    // Stage h = [x | messages] tile (16 x 128) into LDS as bf16, coalesced
#pragma unroll
    for (int i = 0; i < 8; ++i) {
        int idx = i * 32 + lane;        // 0..255
        int row = idx >> 4;             // 0..15
        int q   = idx & 15;             // float4 within 64-wide row
        float4 xv = *(const float4*)(x   + (base + row) * DDIM + q * 4);
        float4 mv = *(const float4*)(msg + (base + row) * DDIM + q * 4);
        unsigned short* p0 = h + row * HDIM + q * 4;
        p0[0] = f2bf(xv.x); p0[1] = f2bf(xv.y); p0[2] = f2bf(xv.z); p0[3] = f2bf(xv.w);
        unsigned short* p1 = h + row * HDIM + 64 + q * 4;
        p1[0] = f2bf(mv.x); p1[1] = f2bf(mv.y); p1[2] = f2bf(mv.z); p1[3] = f2bf(mv.w);
    }
    __syncthreads();

    const int m  = lane & 15;           // A-fragment row
    const int kb = (lane >> 4) * 8;     // A-fragment K sub-block
    const int hi = lane >> 4;           // C-fragment row-half
    const int nl = lane & 15;           // C-fragment column
    const unsigned int* hl = (const unsigned int*)h;

    // ---- GEMM1: hidden[16x128] = h[16x128] @ W1[128x128] ----
    f32x8 acc1[8];
#pragma unroll
    for (int nt = 0; nt < 8; ++nt) acc1[nt] = (f32x8)0.0f;

#pragma unroll
    for (int kc = 0; kc < 4; ++kc) {
        Frag16 a;
#pragma unroll
        for (int j = 0; j < 8; ++j) {
            int K = (j < 4) ? (kb + 2 * j) : (16 + kb + 2 * (j - 4));
            a.u[j] = hl[(m * HDIM + kc * 32 + K) >> 1];
        }
#pragma unroll
        for (int nt = 0; nt < 8; ++nt) {
            Frag16 b;
            const unsigned int* wp = (const unsigned int*)(W1bf + ((kc * 8 + nt) * 32 + lane) * 16);
#pragma unroll
            for (int j = 0; j < 8; ++j) b.u[j] = wp[j];
            acc1[nt] = __builtin_amdgcn_wmma_f32_16x16x32_bf16(
                false, a.v, false, b.v, (short)0, acc1[nt], false, false);
        }
    }
    __syncthreads();

    // bias + ReLU, hidden back to LDS (bf16) for re-fragmentation as A of GEMM2
#pragma unroll
    for (int nt = 0; nt < 8; ++nt) {
        float bias = b1[nt * 16 + nl];
#pragma unroll
        for (int r = 0; r < 8; ++r) {
            float v = acc1[nt][r] + bias;
            v = v > 0.f ? v : 0.f;
            h[(r + hi * 8) * HDIM + nt * 16 + nl] = f2bf(v);
        }
    }
    __syncthreads();

    // ---- GEMM2: f[16x64] = hidden[16x128] @ W2[128x64] ----
    f32x8 acc2[4];
#pragma unroll
    for (int nt = 0; nt < 4; ++nt) acc2[nt] = (f32x8)0.0f;

#pragma unroll
    for (int kc = 0; kc < 4; ++kc) {
        Frag16 a;
#pragma unroll
        for (int j = 0; j < 8; ++j) {
            int K = (j < 4) ? (kb + 2 * j) : (16 + kb + 2 * (j - 4));
            a.u[j] = hl[(m * HDIM + kc * 32 + K) >> 1];
        }
#pragma unroll
        for (int nt = 0; nt < 4; ++nt) {
            Frag16 b;
            const unsigned int* wp = (const unsigned int*)(W2bf + ((kc * 4 + nt) * 32 + lane) * 16);
#pragma unroll
            for (int j = 0; j < 8; ++j) b.u[j] = wp[j];
            acc2[nt] = __builtin_amdgcn_wmma_f32_16x16x32_bf16(
                false, a.v, false, b.v, (short)0, acc2[nt], false, false);
        }
    }

    // epilogue: x += DT*(f - x); e += DT*(-e/TAU + x_new)
    if (active) {
#pragma unroll
        for (int nt = 0; nt < 4; ++nt) {
            float bias = b2[nt * 16 + nl];
#pragma unroll
            for (int r = 0; r < 8; ++r) {
                int row = base + r + hi * 8;
                int col = nt * 16 + nl;
                float f  = acc2[nt][r] + bias;
                float xv = x[row * DDIM + col];
                float xn = xv + DT_F * (f - xv);
                x[row * DDIM + col] = xn;
                float ev = e[row * DDIM + col];
                e[row * DDIM + col] = ev + DT_F * (-ev * INV_TAU + xn);
            }
        }
    }
}

// -------------------------------------------------------------------------
// dw = etaP*<x[src],e[tgt]> - etaM*<x[tgt],e[src]> - lam*w ; 16 lanes/edge
__global__ void edge_plasticity(const int* __restrict__ ei, const float* __restrict__ x,
                                const float* __restrict__ e, float* __restrict__ w,
                                const float* __restrict__ etaP_p, const float* __restrict__ etaM_p) {
    int t = blockIdx.x * blockDim.x + threadIdx.x;
    if (t >= N_EDGES * 16) return;
    int eid = t >> 4, part = t & 15;
    int src = ei[eid], tgt = ei[N_EDGES + eid];
    int c = part * 4;
    float4 xs = *(const float4*)(x + src * DDIM + c);
    float4 et = *(const float4*)(e + tgt * DDIM + c);
    float4 xt = *(const float4*)(x + tgt * DDIM + c);
    float4 es = *(const float4*)(e + src * DDIM + c);
    float p = xs.x * et.x + xs.y * et.y + xs.z * et.z + xs.w * et.w;
    float q = xt.x * es.x + xt.y * es.y + xt.z * es.z + xt.w * es.w;
#pragma unroll
    for (int off = 8; off >= 1; off >>= 1) {
        p += __shfl_xor(p, off, 16);
        q += __shfl_xor(q, off, 16);
    }
    if (part == 0) {
        float wv = w[eid];
        float dw = etaP_p[0] * p - etaM_p[0] * q - LAMBDA_W * wv;
        w[eid] = wv + DT_F * dw;
    }
}

// -------------------------------------------------------------------------
extern "C" void kernel_launch(void* const* d_in, const int* in_sizes, int n_in,
                              void* d_out, int out_size, void* d_ws, size_t ws_size,
                              hipStream_t stream) {
    const float* x_in = (const float*)d_in[0];
    const int*   ei   = (const int*)d_in[1];     // edge_index (2, NE)
    const float* ea   = (const float*)d_in[2];   // edge_attr
    const float* etaP = (const float*)d_in[3];
    const float* etaM = (const float*)d_in[4];
    const float* W1   = (const float*)d_in[5];
    const float* b1   = (const float*)d_in[6];
    const float* W2   = (const float*)d_in[7];
    const float* b2   = (const float*)d_in[8];

    // d_out doubles as the working buffers for x (first) and w (second)
    float* x = (float*)d_out;
    float* w = (float*)d_out + (size_t)N_NODES * DDIM;

    char* ws = (char*)d_ws;
    const size_t node_bytes = (size_t)N_NODES * DDIM * sizeof(float);  // 12.8 MB
    float* e   = (float*)ws;
    float* msg = (float*)(ws + node_bytes);
    unsigned short* W1bf = (unsigned short*)(ws + 2 * node_bytes);
    unsigned short* W2bf = W1bf + 4 * 8 * 512;

    // init working state (deterministic each call; no input mutation)
    hipMemcpyAsync(x, x_in, node_bytes, hipMemcpyDeviceToDevice, stream);
    hipMemcpyAsync(w, ea, (size_t)N_EDGES * sizeof(float), hipMemcpyDeviceToDevice, stream);
    zero_f32<<<(N_NODES * DDIM / 4 + 255) / 256, 256, 0, stream>>>(e, N_NODES * DDIM);
    prep_weights<<<(24576 + 255) / 256, 256, 0, stream>>>(W1, W2, W1bf, W2bf);

    const int edge_threads = N_EDGES * 16;          // 12.8M, blocks of 256 exactly
    for (int s = 0; s < 10; ++s) {
        zero_f32<<<(N_NODES * DDIM / 4 + 255) / 256, 256, 0, stream>>>(msg, N_NODES * DDIM);
        scatter_msgs<<<edge_threads / 256, 256, 0, stream>>>(ei, w, x, msg);
        node_mlp<<<(NTILES + 7) / 8, 256, 0, stream>>>(x, e, msg, W1bf, W2bf, b1, b2);
        edge_plasticity<<<edge_threads / 256, 256, 0, stream>>>(ei, x, e, w, etaP, etaM);
    }
}
